// Attention_73770358276563
// MI455X (gfx1250) — matrix operand order repaired and checked by reading, not compiled, not run on hardware
//
#include <hip/hip_runtime.h>

// ---------------------------------------------------------------------------
// Fused multi-head self-attention for (B,N,T,C)=(4,207,128,64), HIDDEN=128,
// HEADS=8, DH=16.  One workgroup per (b,n) slice; all intermediates in LDS.
// bf16 WMMA (v_wmma_f32_16x16x32_bf16) with fp32 accumulation + fp32 softmax.
// X tile staged via TDM (tensor_load_to_lds + s_wait_tensorcnt).
// LDS pitches padded to kill bank conflicts (64 banks x 4B).
// ---------------------------------------------------------------------------

typedef __bf16 bf16;
typedef __attribute__((ext_vector_type(16))) __bf16 v16bf;
typedef __attribute__((ext_vector_type(4)))  __bf16 bf16x4;
typedef __attribute__((ext_vector_type(8)))  float  v8f;
typedef __attribute__((ext_vector_type(4)))  unsigned int u32x4;
typedef __attribute__((ext_vector_type(8)))  unsigned int u32x8;

#define T_DIM 128
#define C_DIM 64
#define HID   128
#define NHEAD 8
#define DH    16

// padded LDS pitches (elements)
#define QP 136   // bf16 images: Qb, Kb, Vt, Hb, scoreB, WprojT  (272B = 68 dw)
#define SP 132   // f32 scores                                    (528B = 132 dw)
#define XP 72    // bf16 staging: Xb, WtT                         (144B = 36 dw)

__device__ __forceinline__ v8f wmma_bf16(v16bf a, v16bf b, v8f c) {
  return __builtin_amdgcn_wmma_f32_16x16x32_bf16(
      /*neg_a=*/false, a, /*neg_b=*/false, b,
      /*c_mod=*/(short)0, c, /*reuse_a=*/false, /*reuse_b=*/false);
}

// A fragment (16x32 bf16, MxK) from row-major LDS, pitch in elems.
// lane l: row M = m0+(l&15); elems 0..7 -> k = kc+(l>>4)*8+0..7,
//         elems 8..15 -> k = kc+16+(l>>4)*8+0..7.
__device__ __forceinline__ v16bf load_a(const bf16* base, int pitch,
                                        int m0, int kc, int lane) {
  int row = m0 + (lane & 15);
  int kb  = kc + ((lane >> 4) << 3);
  const bf16* p = base + row * pitch + kb;
  v16bf a;
#pragma unroll
  for (int i = 0; i < 8; ++i) { a[i] = p[i]; a[8 + i] = p[16 + i]; }
  return a;
}

// A fragment with only K=0..15 real (upper half of the K=32 chunk zero).
__device__ __forceinline__ v16bf load_a_k16(const bf16* base, int pitch,
                                            int m0, int col0, int lane) {
  int row = m0 + (lane & 15);
  int kb  = (lane >> 4) << 3;   // 0 or 8
  const bf16* p = base + row * pitch + col0 + kb;
  bf16 z = (bf16)0.0f;
  v16bf a;
#pragma unroll
  for (int i = 0; i < 8; ++i) { a[i] = p[i]; a[8 + i] = z; }
  return a;
}

// B fragment (32x16 bf16, KxN) from N-major LDS bt[n][k].
// lane l: col N = n0+(l&15); elems 0..7 -> k = kc+(l>>4)*8+0..7,
//         elems 8..15 -> k = kc+16+(l>>4)*8+0..7.
__device__ __forceinline__ v16bf load_b(const bf16* bt, int pitch,
                                        int n0, int kc, int lane) {
  int n  = n0 + (lane & 15);
  int kb = kc + ((lane >> 4) << 3);
  const bf16* p = bt + n * pitch + kb;
  v16bf b;
#pragma unroll
  for (int i = 0; i < 8; ++i) { b[i] = p[i]; b[8 + i] = p[16 + i]; }
  return b;
}

// B fragment, only K=0..15 real, from row-major src[n][col0+k] (energy path).
__device__ __forceinline__ v16bf load_b_k16(const bf16* src, int pitch,
                                            int n0, int col0, int lane) {
  v16bf b;
  bf16 z = (bf16)0.0f;
  const bf16* p = src + (n0 + (lane & 15)) * pitch + col0 + ((lane >> 4) << 3);
#pragma unroll
  for (int i = 0; i < 8; ++i) { b[i] = p[i]; b[8 + i] = z; }
  return b;
}

// D stores; element (r,lane): M = m0 + r + (lane>>4)*8, N = n0 + (lane&15).
__device__ __forceinline__ void store_d_bf16(bf16* dst, int pitch, int m0,
                                             int n0, v8f d, int lane) {
  int n  = n0 + (lane & 15);
  int mh = m0 + ((lane >> 4) << 3);
#pragma unroll
  for (int r = 0; r < 8; ++r) dst[(mh + r) * pitch + n] = (bf16)d[r];
}

__device__ __forceinline__ void store_d_bf16_T(bf16* dst, int pitch, int m0,
                                               int n0, v8f d, int lane) {
  int n  = n0 + (lane & 15);
  int mh = m0 + ((lane >> 4) << 3);
#pragma unroll
  for (int r = 0; r < 8; ++r) dst[n * pitch + (mh + r)] = (bf16)d[r];
}

__device__ __forceinline__ void store_d_f32(float* dst, int pitch, int m0,
                                            int n0, v8f d, int lane) {
  int n  = n0 + (lane & 15);
  int mh = m0 + ((lane >> 4) << 3);
#pragma unroll
  for (int r = 0; r < 8; ++r) dst[(mh + r) * pitch + n] = d[r];
}

// LDS map (247808 B total, all bases 16B aligned):
//   [0      ,  34816)  Qb   bf16 [128][QP]   | phase0: X f32 staging (32768B)
//   [34816  ,  69632)  Kb   bf16 [128][QP]
//   [69632  , 104448)  Vt   bf16 [128][QP]   (Vt[d][t])
//   [104448 , 139264)  Hb   bf16 [128][QP]
//   [139264 , 212992)  scoresF f32 [128][SP] | phase0: Xb(18432)+WtT(55296)
//   [212992 , 247808)  scoreB bf16 [128][QP] | phase3: WprojT
__global__ __launch_bounds__(256)
void attn_fused_kernel(const float* __restrict__ x,
                       const float* __restrict__ Wq,
                       const float* __restrict__ Wk,
                       const float* __restrict__ Wv,
                       const float* __restrict__ Wproj,
                       float* __restrict__ out) {
  __shared__ __align__(16) char smem[247808];
  bf16*  Qb      = (bf16*)(smem);
  bf16*  Kb      = (bf16*)(smem + 34816);
  bf16*  Vt      = (bf16*)(smem + 69632);
  bf16*  Hb      = (bf16*)(smem + 104448);
  float* scoresF = (float*)(smem + 139264);
  bf16*  Xb      = (bf16*)(smem + 139264);          // aliases scoresF
  bf16*  WtT     = (bf16*)(smem + 157696);          // 3 x [128][XP]
  bf16*  scoreB  = (bf16*)(smem + 212992);
  bf16*  WpT     = (bf16*)(smem + 212992);          // aliases scoreB
  const float* Xstage = (const float*)(smem);       // aliases Qb (phase 0)

  const int tid  = threadIdx.x;
  const int lane = tid & 31;
  const int wave = tid >> 5;           // 0..7 -> m-tile
  const int m0   = wave * 16;
  const int bn   = blockIdx.x;         // 0 .. B*N-1

  const float* xb = x + (size_t)bn * (T_DIM * C_DIM);

  // ---- Phase 0a: TDM DMA of the X tile (32KB f32) into the Qb region ------
  if (wave == 0) {
    unsigned long long ga = (unsigned long long)(const void*)xb;  // [56:0]
    unsigned lds_off = (unsigned)(unsigned long long)(const void*)Xstage;
    // D# group 0: count=1, lds_addr, global_addr, type=2 ("image")
    u32x4 g0;
    g0[0] = 1u;                                    // count=1, user mode
    g0[1] = lds_off;                               // lds_addr
    g0[2] = (unsigned)(ga & 0xFFFFFFFFu);          // global_addr[31:0]
    g0[3] = ((unsigned)(ga >> 32) & 0x01FFFFFFu) | 0x80000000u; // [56:32]|type2
    // D# group 1: data_size=3 (8B), 1-D tensor of 4096 elems, tile=4096
    u32x8 g1;
    g1[0] = 3u << 16;                              // data_size=8B
    g1[1] = (4096u & 0xFFFFu) << 16;               // tensor_dim0[15:0]
    g1[2] = (4096u >> 16) | (1u << 16);            // dim0 hi | tensor_dim1=1
    g1[3] = (4096u << 16);                         // tile_dim0=4096
    g1[4] = 1u;                                    // tile_dim1=1
    g1[5] = 4096u;                                 // tensor_dim0_stride lo
    g1[6] = 0u;
    g1[7] = 0u;
    asm volatile("tensor_load_to_lds %0, %1"
                 :: "s"(g0), "s"(g1) : "memory");
    __builtin_amdgcn_s_wait_tensorcnt(0);
  }

  // ---- Phase 0b: stage W^T (n-major) in bf16 (global -> LDS) --------------
  {
    const float* Ws[3] = {Wq, Wk, Wv};
#pragma unroll
    for (int o = 0; o < 3; ++o) {
      bf16* wt = WtT + o * (HID * XP);
      const float* w = Ws[o];
#pragma unroll 4
      for (int i = tid; i < C_DIM * HID; i += 256) {
        int c = i >> 7;          // row in W (C)
        int n = i & 127;         // col in W (HIDDEN)
        wt[n * XP + c] = (bf16)w[i];
      }
    }
  }
  __syncthreads();               // X staged (TDM) + WtT ready

  // ---- Phase 0c: convert staged X f32 -> bf16 with padded pitch -----------
#pragma unroll 4
  for (int i = tid; i < T_DIM * C_DIM; i += 256) {
    int r = i >> 6, c = i & 63;
    Xb[r * XP + c] = (bf16)Xstage[i];
  }
  __syncthreads();               // Xb ready; Qb region free again

  // ---- Phase 1: Q,K (row-major) and V^T, all bf16 -------------------------
  {
    v16bf xa0 = load_a(Xb, XP, m0, 0,  lane);
    v16bf xa1 = load_a(Xb, XP, m0, 32, lane);
#pragma unroll
    for (int o = 0; o < 3; ++o) {
      const bf16* wt = WtT + o * (HID * XP);
#pragma unroll
      for (int nt = 0; nt < 8; ++nt) {
        v16bf b0 = load_b(wt, XP, nt * 16, 0,  lane);
        v16bf b1 = load_b(wt, XP, nt * 16, 32, lane);
        v8f acc = {};
        acc = wmma_bf16(xa0, b0, acc);
        acc = wmma_bf16(xa1, b1, acc);
        if (o == 0)      store_d_bf16  (Qb, QP, m0, nt * 16, acc, lane);
        else if (o == 1) store_d_bf16  (Kb, QP, m0, nt * 16, acc, lane);
        else             store_d_bf16_T(Vt, QP, m0, nt * 16, acc, lane);
      }
    }
  }
  __syncthreads();

  // ---- Phase 2: per-head attention ---------------------------------------
  const float scale = 0.25f;     // 1/sqrt(DH), DH=16
  for (int h = 0; h < NHEAD; ++h) {
    const int c0 = h * DH;

    // 2a: energy = Q_h @ K_h^T  (K=16 padded to 32)
    {
      v16bf qa = load_a_k16(Qb, QP, m0, c0, lane);
#pragma unroll
      for (int nt = 0; nt < 8; ++nt) {
        v16bf kb = load_b_k16(Kb, QP, nt * 16, c0, lane);
        v8f e = {};
        e = wmma_bf16(qa, kb, e);
        store_d_f32(scoresF, SP, m0, nt * 16, e, lane);
      }
    }
    __syncthreads();

    // 2b: fp32 softmax along rows (vectorized float4), emit bf16 scores
    if (tid < T_DIM) {
      float*  row  = scoresF + tid * SP;
      bf16*   brow = scoreB  + tid * QP;
      float4* r4   = (float4*)row;
      float mx = -3.0e38f;
#pragma unroll 8
      for (int j = 0; j < T_DIM / 4; ++j) {
        float4 v = r4[j];
        mx = fmaxf(mx, fmaxf(fmaxf(v.x, v.y), fmaxf(v.z, v.w)));
      }
      float sum = 0.0f;
#pragma unroll 4
      for (int j = 0; j < T_DIM / 4; ++j) {
        float4 v = r4[j];
        v.x = __expf(scale * (v.x - mx));
        v.y = __expf(scale * (v.y - mx));
        v.z = __expf(scale * (v.z - mx));
        v.w = __expf(scale * (v.w - mx));
        r4[j] = v;
        sum += (v.x + v.y) + (v.z + v.w);
      }
      float inv = 1.0f / sum;
#pragma unroll 8
      for (int j = 0; j < T_DIM / 4; ++j) {
        float4 v = r4[j];
        bf16x4 p;
        p[0] = (bf16)(v.x * inv);
        p[1] = (bf16)(v.y * inv);
        p[2] = (bf16)(v.z * inv);
        p[3] = (bf16)(v.w * inv);
        *(bf16x4*)(brow + j * 4) = p;
      }
    }
    __syncthreads();

    // 2c: head_out = score @ V_h  (M=128, N=16, K=128 -> 4 WMMA)
    {
      v8f ho = {};
#pragma unroll
      for (int kc = 0; kc < T_DIM; kc += 32) {
        v16bf sa = load_a(scoreB, QP, m0, kc, lane);
        v16bf vb = load_b(Vt, QP, c0, kc, lane);
        ho = wmma_bf16(sa, vb, ho);
      }
      store_d_bf16(Hb, QP, m0, c0, ho, lane);
    }
    __syncthreads();
  }

  // ---- Phase 3: out = H @ Wproj ------------------------------------------
#pragma unroll 4
  for (int i = tid; i < HID * HID; i += 256) {
    int r = i >> 7;              // K row
    int n = i & 127;             // N col
    WpT[n * QP + r] = (bf16)Wproj[i];
  }
  __syncthreads();

  {
    float* outp = out + (size_t)bn * (T_DIM * HID);
    v16bf ha[4];
#pragma unroll
    for (int kc = 0; kc < 4; ++kc) ha[kc] = load_a(Hb, QP, m0, kc * 32, lane);
#pragma unroll
    for (int nt = 0; nt < 8; ++nt) {
      v8f acc = {};
#pragma unroll
      for (int kc = 0; kc < 4; ++kc) {
        v16bf wb = load_b(WpT, QP, nt * 16, kc * 32, lane);
        acc = wmma_bf16(ha[kc], wb, acc);
      }
      int n  = nt * 16 + (lane & 15);
      int mh = m0 + ((lane >> 4) << 3);
#pragma unroll
      for (int r = 0; r < 8; ++r) outp[(mh + r) * HID + n] = acc[r];
    }
  }
}

extern "C" void kernel_launch(void* const* d_in, const int* in_sizes, int n_in,
                              void* d_out, int out_size, void* d_ws, size_t ws_size,
                              hipStream_t stream) {
  (void)in_sizes; (void)n_in; (void)d_ws; (void)ws_size; (void)out_size;
  const float* x     = (const float*)d_in[0];
  const float* Wq    = (const float*)d_in[1];
  const float* Wk    = (const float*)d_in[2];
  const float* Wv    = (const float*)d_in[3];
  const float* Wproj = (const float*)d_in[4];
  float* out = (float*)d_out;

  const int B = 4, N = 207;
  dim3 grid(B * N), block(256);
  attn_fused_kernel<<<grid, block, 0, stream>>>(x, Wq, Wk, Wv, Wproj, out);
}
